// Fusion_Layer_61572651155712
// MI455X (gfx1250) — matrix operand order repaired and checked by reading, not compiled, not run on hardware
//
#include <hip/hip_runtime.h>
#include <hip/hip_bf16.h>

#define DEVINL __device__ __forceinline__

typedef __attribute__((ext_vector_type(16))) __bf16        v16bf;
typedef __attribute__((ext_vector_type(8)))  float         v8f;
typedef __attribute__((ext_vector_type(8)))  unsigned int  v8u;

union FragU { v8u u; v16bf b; };

DEVINL float bf2f(unsigned short h) { return __uint_as_float(((unsigned)h) << 16); }
DEVINL unsigned short f2bf(float f) {
  unsigned u = __float_as_uint(f);
  u += 0x7FFFu + ((u >> 16) & 1u);           // round-to-nearest-even
  return (unsigned short)(u >> 16);
}
DEVINL float gelu_f(float x) { return 0.5f * x * (1.0f + erff(x * 0.7071067811865475f)); }

// ---------------------------------------------------------------------------
// WMMA fragment loads (bf16, wave32). A: 16x32 (M x K), B: 32x16 (K x N).
// A layout (ISA 7.12.2): lane M = lane&15; VGPR r holds K pair
//   kb = (r>>2)*16 + (lane>>4)*8 + (r&3)*2
// B layout: lane N = lane&15; VGPR r holds K pair  kb = (r>>2)*16 + (lane>>4)*8 + (r&3)*2
// Both are contiguous dword loads for K-major (NT) operands.
// ---------------------------------------------------------------------------
DEVINL v16bf load_frag_a(const unsigned short* A, long lda, int row, int hh, int k0) {
  FragU f;
  const unsigned short* p = A + (long)row * lda + k0;
#pragma unroll
  for (int r = 0; r < 8; ++r) {
    int kb = ((r >> 2) << 4) + (hh << 3) + ((r & 3) << 1);
    f.u[r] = *(const unsigned int*)(p + kb);
  }
  return f.b;
}
DEVINL v16bf load_frag_b(const unsigned short* Bt, long ldb, int row, int hh, int k0) {
  FragU f;
  const unsigned short* p = Bt + (long)row * ldb + k0;
#pragma unroll
  for (int r = 0; r < 8; ++r) {
    int kb = ((r >> 2) << 4) + (hh << 3) + ((r & 3) << 1);
    f.u[r] = *(const unsigned int*)(p + kb);
  }
  return f.b;
}

// ---------------------------------------------------------------------------
// Generic batched NT GEMM:  C[z](MxN) = act( (A[z](MxK) * Bt[z](NxK)^T)*alpha
//                                            [+res]*resscale )
// z decomposed as z0=z/zdiv, z1=z%zdiv with independent strides per operand.
// Block: 256 thr = 8 waves arranged 4(M) x 2(N); wave tile 32x64
// (2 A-frags x 4 B-frags -> 8 v_wmma per 32-K step); block tile 128x128.
// ---------------------------------------------------------------------------
__global__ __launch_bounds__(256) void gemm_bf16_nt(
    const unsigned short* __restrict__ A, const unsigned short* __restrict__ Bt,
    int M, int N, int K, long lda, long ldb, long ldc,
    long sA0, long sA1, long sB0, long sB1, long sC0, long sC1, int zdiv,
    float alpha, const float* __restrict__ res, float resscale, int act, int transC,
    unsigned short* __restrict__ Cb, float* __restrict__ Cf)
{
  int z = blockIdx.z;
  int z0 = z / zdiv, z1 = z % zdiv;
  A  += z0 * sA0 + z1 * sA1;
  Bt += z0 * sB0 + z1 * sB1;
  long coff = z0 * sC0 + z1 * sC1;

  int lane = threadIdx.x & 31;
  int wave = threadIdx.x >> 5;
  int wm = wave >> 1, wn = wave & 1;
  int m0 = blockIdx.y * 128 + wm * 32;
  int n0 = blockIdx.x * 128 + wn * 64;
  int hh = lane >> 4, l16 = lane & 15;

  int ma[2], na[4];
#pragma unroll
  for (int mi = 0; mi < 2; ++mi) {
    int r = m0 + mi * 16 + l16; if (r > M - 1) r = M - 1; ma[mi] = r;
  }
#pragma unroll
  for (int ni = 0; ni < 4; ++ni) {
    int r = n0 + ni * 16 + l16; if (r > N - 1) r = N - 1; na[ni] = r;
  }

  v8f acc[2][4] = {};
  for (int k0 = 0; k0 < K; k0 += 32) {
    if (k0 + 32 < K) {                         // -> global_prefetch_b8
      __builtin_prefetch(A  + (long)ma[0] * lda + k0 + 32, 0, 1);
      __builtin_prefetch(Bt + (long)na[0] * ldb + k0 + 32, 0, 1);
    }
    v16bf af[2], bf[4];
#pragma unroll
    for (int mi = 0; mi < 2; ++mi) af[mi] = load_frag_a(A, lda, ma[mi], hh, k0);
#pragma unroll
    for (int ni = 0; ni < 4; ++ni) bf[ni] = load_frag_b(Bt, ldb, na[ni], hh, k0);
#pragma unroll
    for (int mi = 0; mi < 2; ++mi)
#pragma unroll
      for (int ni = 0; ni < 4; ++ni)
        acc[mi][ni] = __builtin_amdgcn_wmma_f32_16x16x32_bf16(
            false, af[mi], false, bf[ni], (short)0, acc[mi][ni], false, false);
  }

#pragma unroll
  for (int mi = 0; mi < 2; ++mi) {
#pragma unroll
    for (int ni = 0; ni < 4; ++ni) {
      int nn = n0 + ni * 16 + l16;
#pragma unroll
      for (int r = 0; r < 8; ++r) {
        int mm = m0 + mi * 16 + r + hh * 8;    // ISA 16x16 f32 C/D layout
        if (mm < M && nn < N) {
          float v = acc[mi][ni][r] * alpha;
          long idx = transC ? (coff + (long)nn * ldc + mm)
                            : (coff + (long)mm * ldc + nn);
          if (res) v = (v + res[idx]) * resscale;
          if (act) v = gelu_f(v);
          if (Cf) Cf[idx] = v;
          if (Cb) Cb[idx] = f2bf(v);
        }
      }
    }
  }
}

// ---------------------------------------------------------------------------
// LayerNorm over last dim D (=512), one block per row. Optional bf16 copy.
// ---------------------------------------------------------------------------
__global__ __launch_bounds__(256) void layernorm_k(
    const float* __restrict__ x, const float* __restrict__ g, const float* __restrict__ b,
    float* __restrict__ outf, unsigned short* __restrict__ outb, int D)
{
  long row = blockIdx.x;
  const float* xr = x + row * (long)D;
  float s = 0.f, s2 = 0.f;
  for (int i = threadIdx.x; i < D; i += 256) { float v = xr[i]; s += v; s2 += v * v; }
  __shared__ float sh[16];
  for (int o = 16; o; o >>= 1) { s += __shfl_down(s, o, 32); s2 += __shfl_down(s2, o, 32); }
  int lane = threadIdx.x & 31, wv = threadIdx.x >> 5;
  if (lane == 0) { sh[wv] = s; sh[8 + wv] = s2; }
  __syncthreads();
  if (threadIdx.x == 0) {
    float a = 0.f, c = 0.f;
    for (int w = 0; w < 8; ++w) { a += sh[w]; c += sh[8 + w]; }
    sh[0] = a; sh[8] = c;
  }
  __syncthreads();
  float mean = sh[0] / (float)D;
  float var  = sh[8] / (float)D - mean * mean;
  float inv  = rsqrtf(var + 1e-5f);
  for (int i = threadIdx.x; i < D; i += 256) {
    float v = (xr[i] - mean) * inv * g[i] + b[i];
    if (outf) outf[row * (long)D + i] = v;
    if (outb) outb[row * (long)D + i] = f2bf(v);
  }
}

// ---------------------------------------------------------------------------
// Row softmax on bf16, in place. One block per row of length L.
// ---------------------------------------------------------------------------
__global__ __launch_bounds__(256) void softmax_bf16_k(unsigned short* att, int L)
{
  long row = blockIdx.x;
  unsigned short* p = att + row * (long)L;
  int lane = threadIdx.x & 31, wv = threadIdx.x >> 5;
  __shared__ float shm[8];
  __shared__ float shs[8];
  float mx = -3.0e38f;
  for (int i = threadIdx.x; i < L; i += 256) mx = fmaxf(mx, bf2f(p[i]));
  for (int o = 16; o; o >>= 1) mx = fmaxf(mx, __shfl_down(mx, o, 32));
  if (lane == 0) shm[wv] = mx;
  __syncthreads();
  if (threadIdx.x == 0) { float m = shm[0]; for (int w = 1; w < 8; ++w) m = fmaxf(m, shm[w]); shm[0] = m; }
  __syncthreads();
  mx = shm[0];
  float s = 0.f;
  for (int i = threadIdx.x; i < L; i += 256) s += __expf(bf2f(p[i]) - mx);
  for (int o = 16; o; o >>= 1) s += __shfl_down(s, o, 32);
  if (lane == 0) shs[wv] = s;
  __syncthreads();
  if (threadIdx.x == 0) { float t = 0.f; for (int w = 0; w < 8; ++w) t += shs[w]; shs[0] = t; }
  __syncthreads();
  float inv = 1.0f / shs[0];
  for (int i = threadIdx.x; i < L; i += 256) p[i] = f2bf(__expf(bf2f(p[i]) - mx) * inv);
}

// ---------------------------------------------------------------------------
// Batched bf16 transpose: in (z,R,C) -> out (z,C,R), 32x32 LDS tiles.
// ---------------------------------------------------------------------------
__global__ void transpose_bf16_k(const unsigned short* __restrict__ in,
                                 unsigned short* __restrict__ out, int R, int C)
{
  __shared__ unsigned short tile[32][33];
  long zo = (long)blockIdx.z * R * C;
  int c0 = blockIdx.x * 32, r0 = blockIdx.y * 32;
  for (int i = threadIdx.y; i < 32; i += 8) {
    int r = r0 + i, c = c0 + threadIdx.x;
    tile[i][threadIdx.x] = (r < R && c < C) ? in[zo + (long)r * C + c] : (unsigned short)0;
  }
  __syncthreads();
  for (int i = threadIdx.y; i < 32; i += 8) {
    int r = r0 + (int)threadIdx.x, c = c0 + i;
    if (c < C && r < R) out[zo + (long)c * R + r] = tile[threadIdx.x][i];
  }
}

// ---------------------------------------------------------------------------
// float -> bf16 convert
// ---------------------------------------------------------------------------
__global__ __launch_bounds__(256) void f2bf_k(const float* __restrict__ in,
                                              unsigned short* __restrict__ out, long n)
{
  long i = (long)blockIdx.x * 256 + threadIdx.x;
  if (i < n) out[i] = f2bf(in[i]);
}

// ---------------------------------------------------------------------------
// Fused detail path (7x7 conv + bn + 1x1 convs + gelu) and combine with glob:
//   y = (glob + detail(ln)) / 2 ; write f32 and bf16.
// Image is (B, 1, SQ=1024, D=512), zero padding 3.
// ---------------------------------------------------------------------------
__global__ __launch_bounds__(256) void detail_combine_k(
    const float* __restrict__ ln, const float* __restrict__ glob,
    const float* __restrict__ c7,  const float* __restrict__ bn1g, const float* __restrict__ bn1b,
    const float* __restrict__ c1,  const float* __restrict__ bn2g, const float* __restrict__ bn2b,
    const float* __restrict__ c2,  const float* __restrict__ bn3g, const float* __restrict__ bn3b,
    float* __restrict__ yf, unsigned short* __restrict__ yb)
{
  const float IS = 0.9999950000374997f;  // 1/sqrt(1+1e-5)
  long idx = (long)blockIdx.x * 256 + threadIdx.x;
  if (idx >= 8L * 1024 * 512) return;
  int  c = (int)(idx % 512);
  long t = idx / 512;
  int  s = (int)(t % 1024);
  long b = t / 1024;
  const float* img = ln + b * (1024L * 512);
  float acc = 0.f;
  for (int dy = -3; dy <= 3; ++dy) {
    int ss = s + dy;
    if (ss < 0 || ss >= 1024) continue;
    for (int dx = -3; dx <= 3; ++dx) {
      int cc = c + dx;
      if (cc < 0 || cc >= 512) continue;
      acc += img[(long)ss * 512 + cc] * c7[(dy + 3) * 7 + (dx + 3)];
    }
  }
  float t1 = acc * (bn1g[0] * IS) + bn1b[0];
  float det = 0.f;
#pragma unroll
  for (int ch = 0; ch < 32; ++ch) {
    float v = t1 * c1[ch];
    v = v * (bn2g[ch] * IS) + bn2b[ch];
    det += gelu_f(v) * c2[ch];
  }
  det = det * (bn3g[0] * IS) + bn3b[0];
  float y = (glob[idx] + det) * 0.5f;
  yf[idx] = y;
  yb[idx] = f2bf(y);
}

// ---------------------------------------------------------------------------
// ysum = gelu(conv7x7(stack(y1,y2,y3)))  (3 in-ch -> 1 out-ch, pad 3)
// ---------------------------------------------------------------------------
__global__ __launch_bounds__(256) void cbconv_k(
    const float* __restrict__ y1, const float* __restrict__ y2, const float* __restrict__ y3,
    const float* __restrict__ wcb, float* __restrict__ outf, unsigned short* __restrict__ outb)
{
  long idx = (long)blockIdx.x * 256 + threadIdx.x;
  if (idx >= 8L * 1024 * 512) return;
  int  c = (int)(idx % 512);
  long t = idx / 512;
  int  s = (int)(t % 1024);
  long b = t / 1024;
  const float* imgs[3] = { y1 + b * (1024L * 512), y2 + b * (1024L * 512), y3 + b * (1024L * 512) };
  float acc = 0.f;
  for (int ch = 0; ch < 3; ++ch) {
    const float* img = imgs[ch];
    const float* wk  = wcb + ch * 49;
    for (int dy = -3; dy <= 3; ++dy) {
      int ss = s + dy;
      if (ss < 0 || ss >= 1024) continue;
      for (int dx = -3; dx <= 3; ++dx) {
        int cc = c + dx;
        if (cc < 0 || cc >= 512) continue;
        acc += img[(long)ss * 512 + cc] * wk[(dy + 3) * 7 + (dx + 3)];
      }
    }
  }
  float v = gelu_f(acc);
  outf[idx] = v;
  outb[idx] = f2bf(v);
}

// ---------------------------------------------------------------------------
// Host-side GEMM launcher (block tile 128x128)
// ---------------------------------------------------------------------------
static inline void gemmL(hipStream_t st, const void* A, const void* Bt,
                         int M, int N, int K, long lda, long ldb, long ldc,
                         long sA0, long sA1, long sB0, long sB1, long sC0, long sC1,
                         int zdiv, int Z, float alpha,
                         const float* res, float resscale, int act, int transC,
                         void* Cb, void* Cf)
{
  dim3 g((unsigned)((N + 127) / 128), (unsigned)((M + 127) / 128), (unsigned)Z);
  gemm_bf16_nt<<<g, 256, 0, st>>>((const unsigned short*)A, (const unsigned short*)Bt,
      M, N, K, lda, ldb, ldc, sA0, sA1, sB0, sB1, sC0, sC1, zdiv, alpha, res, resscale,
      act, transC, (unsigned short*)Cb, (float*)Cf);
}

extern "C" void kernel_launch(void* const* d_in, const int* in_sizes, int n_in,
                              void* d_out, int out_size, void* d_ws, size_t ws_size,
                              hipStream_t stream)
{
  (void)in_sizes; (void)n_in; (void)out_size; (void)ws_size;
  const long NS = 8L * 1024 * 512;
  const float* xin[3] = { (const float*)d_in[0], (const float*)d_in[1], (const float*)d_in[2] };
  float* out = (float*)d_out;

  // ---- workspace carving ----
  char* w = (char*)d_ws;
  auto carve = [&](size_t bytes) -> char* {
    char* p = w; w += (bytes + 255) & ~(size_t)255; return p;
  };
  float*          YF    = (float*)carve(3 * NS * 4);            // persistent y1..y3 f32
  unsigned short* YB    = (unsigned short*)carve(3 * NS * 2);   // persistent y1..y3 bf16
  float*          YSUMF = (float*)carve(NS * 4);
  unsigned short* YSUMB = (unsigned short*)carve(NS * 2);
  unsigned short* Wfc1 = (unsigned short*)carve(128L * 512 * 2);
  unsigned short* Wfc2 = (unsigned short*)carve(512L * 128 * 2);
  unsigned short* Wslq = (unsigned short*)carve(512L * 512 * 2);
  unsigned short* Wslk = (unsigned short*)carve(512L * 512 * 2);
  unsigned short* Wslf = (unsigned short*)carve(512L * 512 * 2);
  unsigned short* Wdmq = (unsigned short*)carve(512L * 1024 * 2);
  unsigned short* Wdmk = (unsigned short*)carve(512L * 1024 * 2);
  unsigned short* Wdmf = (unsigned short*)carve(1024L * 512 * 2);
  unsigned short* Ww1  = (unsigned short*)carve(2048L * 512 * 2);
  unsigned short* Ww2  = (unsigned short*)carve(512L * 2048 * 2);
  char* scrbase = w;                                            // per-branch scratch
  float*          LNF  = (float*)carve(NS * 4);
  unsigned short* LNB  = (unsigned short*)carve(NS * 2);
  unsigned short* XTT  = (unsigned short*)carve(8L * 128 * 1024 * 2);   // (b,128,1024)
  unsigned short* QD   = (unsigned short*)carve(1024L * 512 * 2);       // (b*128,512)
  unsigned short* KD   = (unsigned short*)carve(1024L * 512 * 2);
  unsigned short* QDT  = (unsigned short*)carve(1024L * 512 * 2);       // (b,512,128)
  unsigned short* ATTD = (unsigned short*)carve(64L * 128 * 128 * 2);   // (b,h,128,128)
  unsigned short* ODM  = (unsigned short*)carve(1024L * 512 * 2);
  unsigned short* YDMT = (unsigned short*)carve(8L * 1024 * 128 * 2);   // (b,1024,128)
  float*          YSLF = (float*)carve(NS * 4);
  unsigned short* YSLB = (unsigned short*)carve(NS * 2);
  unsigned short* QS   = (unsigned short*)carve(NS * 2);
  unsigned short* KS   = (unsigned short*)carve(NS * 2);
  unsigned short* QST  = (unsigned short*)carve(NS * 2);                // (b,512,1024)
  unsigned short* ATTS = (unsigned short*)carve(8L * 1024 * 1024 * 2);  // (h,1024,1024) per batch
  unsigned short* OSL  = (unsigned short*)carve(NS * 2);
  float*          GLOB = (float*)carve(NS * 4);
  // FFN scratch overlaps dead branch scratch
  unsigned short* HB    = (unsigned short*)scrbase;                          // 8192x2048 bf16
  float*          PRELN = (float*)(scrbase + (size_t)48 * 1024 * 1024);      // 8192x512 f32

  // ---- convert weights to bf16 ----
  auto cvt = [&](int idx, unsigned short* dst, long n) {
    f2bf_k<<<dim3((unsigned)((n + 255) / 256)), 256, 0, stream>>>((const float*)d_in[idx], dst, n);
  };
  cvt(10, Wfc1, 128L * 512);  cvt(11, Wfc2, 512L * 128);
  cvt(12, Wslq, 512L * 512);  cvt(13, Wslk, 512L * 512);  cvt(14, Wslf, 512L * 512);
  cvt(15, Wdmq, 512L * 1024); cvt(16, Wdmk, 512L * 1024); cvt(17, Wdmf, 1024L * 512);
  cvt(46, Ww1, 2048L * 512);  cvt(47, Ww2, 512L * 2048);

  for (int br = 0; br < 3; ++br) {
    layernorm_k<<<8192, 256, 0, stream>>>(xin[br], (const float*)d_in[4 + 2 * br],
        (const float*)d_in[5 + 2 * br], LNF, LNB, 512);

    // xt^T[b] (128x1024): C[f][s] = fc1[f,:] . ln[b][s,:]
    gemmL(stream, Wfc1, LNB, 128, 1024, 512, 512, 512, 1024,
          0, 0, 1024L * 512, 0, 128L * 1024, 0, 1, 8, 1.0f, nullptr, 0.f, 0, 0, XTT, nullptr);
    // Q = xt @ dm_Wq^T  (V == Q by reference construction); K = xt @ dm_Wk^T
    gemmL(stream, XTT, Wdmq, 1024, 512, 1024, 1024, 1024, 512,
          0, 0, 0, 0, 0, 0, 1, 1, 1.0f, nullptr, 0.f, 0, 0, QD, nullptr);
    gemmL(stream, XTT, Wdmk, 1024, 512, 1024, 1024, 1024, 512,
          0, 0, 0, 0, 0, 0, 1, 1, 1.0f, nullptr, 0.f, 0, 0, KD, nullptr);
    transpose_bf16_k<<<dim3(512 / 32, 128 / 32, 8), dim3(32, 8), 0, stream>>>(QD, QDT, 128, 512);
    // scores (b,h,128,128) = Q K^T / 8
    gemmL(stream, QD, KD, 128, 128, 64, 512, 512, 128,
          128L * 512, 64, 128L * 512, 64, 8L * 128 * 128, 128L * 128, 8, 64,
          0.125f, nullptr, 0.f, 0, 0, ATTD, nullptr);
    softmax_bf16_k<<<8192, 256, 0, stream>>>(ATTD, 128);
    // out_dm (b,128,512) = att @ V   (V^T rows from QDT)
    gemmL(stream, ATTD, QDT, 128, 64, 128, 128, 128, 512,
          8L * 128 * 128, 128L * 128, 512L * 128, 64L * 128, 128L * 512, 64, 8, 64,
          1.0f, nullptr, 0.f, 0, 0, ODM, nullptr);
    // y_dm^T (b,1024,128) via transposed store: C[f][s] -> T[s][f]
    gemmL(stream, ODM, Wdmf, 128, 1024, 512, 512, 512, 128,
          128L * 512, 0, 0, 0, 1024L * 128, 0, 1, 8, 1.0f, nullptr, 0.f, 0, 1, YDMT, nullptr);
    // y = (y_dm^T @ fc2^T + ln)/2  -> f32 + bf16
    gemmL(stream, YDMT, Wfc2, 8192, 512, 128, 128, 128, 512,
          0, 0, 0, 0, 0, 0, 1, 1, 1.0f, LNF, 0.5f, 0, 0, YSLB, YSLF);

    // sl self-attention over seq=1024 (V == Q again)
    gemmL(stream, YSLB, Wslq, 8192, 512, 512, 512, 512, 512,
          0, 0, 0, 0, 0, 0, 1, 1, 1.0f, nullptr, 0.f, 0, 0, QS, nullptr);
    gemmL(stream, YSLB, Wslk, 8192, 512, 512, 512, 512, 512,
          0, 0, 0, 0, 0, 0, 1, 1, 1.0f, nullptr, 0.f, 0, 0, KS, nullptr);
    transpose_bf16_k<<<dim3(512 / 32, 1024 / 32, 8), dim3(32, 8), 0, stream>>>(QS, QST, 1024, 512);
    for (int b = 0; b < 8; ++b) {
      const unsigned short* Qb = QS + (long)b * 1024 * 512;
      const unsigned short* Kb = KS + (long)b * 1024 * 512;
      gemmL(stream, Qb, Kb, 1024, 1024, 64, 512, 512, 1024,
            0, 64, 0, 64, 0, 1024L * 1024, 8, 8, 0.125f, nullptr, 0.f, 0, 0, ATTS, nullptr);
      softmax_bf16_k<<<8192, 256, 0, stream>>>(ATTS, 1024);
      gemmL(stream, ATTS, QST + (long)b * 512 * 1024, 1024, 64, 1024, 1024, 1024, 512,
            0, 1024L * 1024, 0, 64L * 1024, 0, 64, 8, 8,
            1.0f, nullptr, 0.f, 0, 0, OSL + (long)b * 1024 * 512, nullptr);
    }
    // glob = (out_sl @ sl_fc^T + y)/2
    gemmL(stream, OSL, Wslf, 8192, 512, 512, 512, 512, 512,
          0, 0, 0, 0, 0, 0, 1, 1, 1.0f, YSLF, 0.5f, 0, 0, nullptr, GLOB);

    int dp = 18 + 9 * br;   // det params: c7,bn1g,bn1b,c1,bn2g,bn2b,c2,bn3g,bn3b
    detail_combine_k<<<dim3((unsigned)((NS + 255) / 256)), 256, 0, stream>>>(
        LNF, GLOB,
        (const float*)d_in[dp + 0], (const float*)d_in[dp + 1], (const float*)d_in[dp + 2],
        (const float*)d_in[dp + 3], (const float*)d_in[dp + 4], (const float*)d_in[dp + 5],
        (const float*)d_in[dp + 6], (const float*)d_in[dp + 7], (const float*)d_in[dp + 8],
        YF + (long)br * NS, YB + (long)br * NS);
  }

  cbconv_k<<<dim3((unsigned)((NS + 255) / 256)), 256, 0, stream>>>(
      YF, YF + NS, YF + 2 * NS, (const float*)d_in[45], YSUMF, YSUMB);

  for (int j = 0; j < 4; ++j) {
    const unsigned short* tb = (j < 3) ? (YB + (long)j * NS) : YSUMB;
    const float*          tf = (j < 3) ? (YF + (long)j * NS) : YSUMF;
    gemmL(stream, tb, Ww1, 8192, 2048, 512, 512, 512, 2048,
          0, 0, 0, 0, 0, 0, 1, 1, 1.0f, nullptr, 0.f, 1, 0, HB, nullptr);       // gelu
    gemmL(stream, HB, Ww2, 8192, 512, 2048, 2048, 2048, 512,
          0, 0, 0, 0, 0, 0, 1, 1, 1.0f, tf, 1.0f, 0, 0, nullptr, PRELN);        // + residual
    layernorm_k<<<8192, 256, 0, stream>>>(PRELN, (const float*)d_in[48 + 2 * j],
        (const float*)d_in[49 + 2 * j], out + (long)j * NS, nullptr, 512);
  }
}
